// BayesPredictor_90323162235656
// MI455X (gfx1250) — matrix operand order repaired and checked by reading, not compiled, not run on hardware
//
#include <hip/hip_runtime.h>
#include <hip/hip_bf16.h>

typedef __bf16 bf16;
typedef __attribute__((ext_vector_type(4)))  bf16  v4bf;
typedef __attribute__((ext_vector_type(8)))  bf16  v8bf;
typedef __attribute__((ext_vector_type(16))) bf16  v16bf;
typedef __attribute__((ext_vector_type(8)))  float v8f;

#define B_  256
#define N_  16384
#define D_  3072
#define KT  64   // K-chunk per pipeline stage
#define LDK 72   // padded LDS row stride in bf16 elements (144 B, 16B-aligned)

// ---------------- CDNA5 async global->LDS copy (ASYNCcnt path) ------------
#if defined(__has_builtin)
#if __has_builtin(__builtin_amdgcn_global_load_async_to_lds_b128)
#define HAVE_ASYNC_LDS 1
#endif
#endif
#ifndef HAVE_ASYNC_LDS
#define HAVE_ASYNC_LDS 0
#endif

#if HAVE_ASYNC_LDS
typedef int v4i_ __attribute__((ext_vector_type(4)));
typedef __attribute__((address_space(1))) v4i_* gas_p;   // global int4*
typedef __attribute__((address_space(3))) v4i_* lds_p;   // LDS int4*
__device__ __forceinline__ void async_copy16(const bf16* g, bf16* l) {
  __builtin_amdgcn_global_load_async_to_lds_b128(
      (gas_p)(size_t)g, (lds_p)(unsigned)(size_t)l, 0, 0);
}
__device__ __forceinline__ void wait_async0() {
#if __has_builtin(__builtin_amdgcn_s_wait_asynccnt)
  __builtin_amdgcn_s_wait_asynccnt(0);
#else
  asm volatile("s_wait_asynccnt 0x0" ::: "memory");
#endif
}
#endif

// ------------------------------------------------------------------
// Per-sample cosine schedule parameters.
__global__ __launch_bounds__(256) void k_params(const int* __restrict__ ts,
                                                float* __restrict__ sqa,
                                                float* __restrict__ rsv,
                                                float* __restrict__ c1,
                                                float* __restrict__ c2,
                                                float* __restrict__ varr) {
  int b = threadIdx.x;
  if (b >= B_) return;
  const float s = 0.008f;
  const float HPI = 1.57079632679489662f;
  float t   = (float)ts[b];
  float c   = cosf(((t / 1000.0f) + s) / (1.0f + s) * HPI);
  float f   = c * c;
  float c0v = cosf(s / (1.0f + s) * HPI);
  float f0  = c0v * c0v;
  float alpha = fminf(fmaxf(f / f0, 1e-5f), 1.0f - 1e-5f);
  float var   = 1.0f - alpha;
  sqa[b]  = sqrtf(alpha);
  rsv[b]  = rsqrtf(var);
  varr[b] = var;
  c1[b] = sqrtf(alpha) / var;       // coeff for xy in logit
  c2[b] = -0.5f * alpha / var;      // coeff for ||y||^2 in logit
}

// ------------------------------------------------------------------
// Convert inputs to bf16, compute ||x_b||^2 and constant logit term c0.
__global__ __launch_bounds__(256) void k_prep_x(const float* __restrict__ x,
                                                bf16* __restrict__ xbf,
                                                const float* __restrict__ varr,
                                                float* __restrict__ c0) {
  int b = blockIdx.x, tid = threadIdx.x;
  __shared__ float red[256];
  const float4* row4 = (const float4*)(x + (size_t)b * D_);
  bf16* orow = xbf + (size_t)b * D_;
  float acc = 0.f;
  for (int i = tid; i < D_ / 4; i += 256) {
    float4 v = row4[i];
    v4bf o; o[0] = (bf16)v.x; o[1] = (bf16)v.y; o[2] = (bf16)v.z; o[3] = (bf16)v.w;
    *(v4bf*)(orow + i * 4) = o;
    acc += v.x * v.x + v.y * v.y + v.z * v.z + v.w * v.w;
  }
  red[tid] = acc; __syncthreads();
  for (int s = 128; s > 0; s >>= 1) {
    if (tid < s) red[tid] += red[tid + s];
    __syncthreads();
  }
  if (tid == 0) {
    float var = varr[b];
    c0[b] = -(0.5f * (float)D_ * logf(var) + 0.5f / var * red[0]);
  }
}

// ------------------------------------------------------------------
// Convert data_batch to bf16 and compute ||y_n||^2 (one block per row).
__global__ __launch_bounds__(256) void k_prep_y(const float* __restrict__ y,
                                                bf16* __restrict__ ybf,
                                                float* __restrict__ ysq) {
  int n = blockIdx.x, tid = threadIdx.x;
  __shared__ float red[256];
  const float4* row4 = (const float4*)(y + (size_t)n * D_);
  bf16* orow = ybf + (size_t)n * D_;
  float acc = 0.f;
  for (int i = tid; i < D_ / 4; i += 256) {
    float4 v = row4[i];
    v4bf o; o[0] = (bf16)v.x; o[1] = (bf16)v.y; o[2] = (bf16)v.z; o[3] = (bf16)v.w;
    *(v4bf*)(orow + i * 4) = o;
    acc += v.x * v.x + v.y * v.y + v.z * v.z + v.w * v.w;
  }
  red[tid] = acc; __syncthreads();
  for (int s = 128; s > 0; s >>= 1) {
    if (tid < s) red[tid] += red[tid + s];
    __syncthreads();
  }
  if (tid == 0) ysq[n] = red[0];
}

// ------------------------------------------------------------------
// GEMM1 (NT): logits[b,n] = c0[b] + c1[b] * <x_b, y_n> + c2[b] * ||y_n||^2
// WG tile 64(M) x 128(N), 8 waves as 4x2, each wave 16x64.
// Double-buffered LDS, one barrier per KT=64 step, 8 WMMA per step per wave.
__global__ __launch_bounds__(256) void k_gemm1(const bf16* __restrict__ xbf,
                                               const bf16* __restrict__ ybf,
                                               const float* __restrict__ ysq,
                                               const float* __restrict__ c0,
                                               const float* __restrict__ c1,
                                               const float* __restrict__ c2,
                                               float* __restrict__ logits) {
  __shared__ bf16 As[2][64 * LDK];
  __shared__ bf16 Bs[2][128 * LDK];
  const int tid  = threadIdx.x;
  const int m0   = blockIdx.y * 64;
  const int n0   = blockIdx.x * 128;
  const int lane = tid & 31, wave = tid >> 5;
  const int wm = wave >> 1, wn = wave & 1;
  const int half = lane >> 4, mm = lane & 15;

  // Copy geometry: 16B chunks; 8 chunks per 64-wide K row.
  int aRow[2], aCol[2], bRow[4], bCol[4];
#pragma unroll
  for (int q = 0; q < 2; ++q) { int c = tid + q * 256; aRow[q] = c >> 3; aCol[q] = (c & 7) * 8; }
#pragma unroll
  for (int q = 0; q < 4; ++q) { int c = tid + q * 256; bRow[q] = c >> 3; bCol[q] = (c & 7) * 8; }

  v8f acc[4] = {};
#if !HAVE_ASYNC_LDS
  v8bf ra[2], rb[4];
#endif

  auto issueCopy = [&](int buf, int k0) {
#if HAVE_ASYNC_LDS
#pragma unroll
    for (int q = 0; q < 2; ++q)
      async_copy16(&xbf[(size_t)(m0 + aRow[q]) * D_ + k0 + aCol[q]],
                   &As[buf][aRow[q] * LDK + aCol[q]]);
#pragma unroll
    for (int q = 0; q < 4; ++q)
      async_copy16(&ybf[(size_t)(n0 + bRow[q]) * D_ + k0 + bCol[q]],
                   &Bs[buf][bRow[q] * LDK + bCol[q]]);
#endif
    (void)buf; (void)k0;
  };
#if !HAVE_ASYNC_LDS
  auto loadStage = [&](int k0) {
#pragma unroll
    for (int q = 0; q < 2; ++q)
      ra[q] = *(const v8bf*)&xbf[(size_t)(m0 + aRow[q]) * D_ + k0 + aCol[q]];
#pragma unroll
    for (int q = 0; q < 4; ++q)
      rb[q] = *(const v8bf*)&ybf[(size_t)(n0 + bRow[q]) * D_ + k0 + bCol[q]];
  };
  auto storeStage = [&](int buf) {
#pragma unroll
    for (int q = 0; q < 2; ++q) *(v8bf*)&As[buf][aRow[q] * LDK + aCol[q]] = ra[q];
#pragma unroll
    for (int q = 0; q < 4; ++q) *(v8bf*)&Bs[buf][bRow[q] * LDK + bCol[q]] = rb[q];
  };
#endif

  auto compute = [&](int buf) {
#pragma unroll
    for (int kh = 0; kh < 2; ++kh) {
      union { v16bf v; v8bf h[2]; } af;
      const bf16* ap = &As[buf][(wm * 16 + mm) * LDK + kh * 32 + half * 8];
      af.h[0] = *(const v8bf*)ap;
      af.h[1] = *(const v8bf*)(ap + 16);
#pragma unroll
      for (int j = 0; j < 4; ++j) {
        union { v16bf v; v8bf h[2]; } bfr;
        const bf16* bp = &Bs[buf][(wn * 64 + j * 16 + mm) * LDK + kh * 32 + half * 8];
        bfr.h[0] = *(const v8bf*)bp;
        bfr.h[1] = *(const v8bf*)(bp + 16);
        acc[j] = __builtin_amdgcn_wmma_f32_16x16x32_bf16(false, af.v, false, bfr.v,
                                                         (short)0, acc[j], false, false);
      }
    }
  };

  const int nK = D_ / KT;
  // Prologue: fill stage 0.
#if HAVE_ASYNC_LDS
  issueCopy(0, 0);
  wait_async0();
#else
  loadStage(0);
  storeStage(0);
#endif
  __syncthreads();

  for (int i = 0; i < nK; ++i) {
    const int cur = i & 1, nxt = cur ^ 1;
    const int k1 = (i + 1) * KT;
#if HAVE_ASYNC_LDS
    if (i + 1 < nK) issueCopy(nxt, k1);   // overlaps with compute below
    compute(cur);
    wait_async0();
#else
    if (i + 1 < nK) loadStage(k1);        // global loads overlap compute
    compute(cur);
    if (i + 1 < nK) storeStage(nxt);
#endif
    __syncthreads();
  }

  // Epilogue: C layout — VGPR i holds M = i + 8*half, column = lane&15.
#pragma unroll
  for (int j = 0; j < 4; ++j) {
    int col = n0 + wn * 64 + j * 16 + mm;
    float yv = ysq[col];
#pragma unroll
    for (int i = 0; i < 8; ++i) {
      int row = m0 + wm * 16 + half * 8 + i;
      logits[(size_t)row * N_ + col] = c0[row] + c1[row] * acc[j][i] + c2[row] * yv;
    }
  }
}

// ------------------------------------------------------------------
// Row softmax: write un-normalized exp weights as bf16; store row sums.
__global__ __launch_bounds__(256) void k_softmax(const float* __restrict__ logits,
                                                 bf16* __restrict__ pbf,
                                                 float* __restrict__ rowsum) {
  int b = blockIdx.x, tid = threadIdx.x;
  __shared__ float red[256];
  const float4* row4 = (const float4*)(logits + (size_t)b * N_);
  bf16* orow = pbf + (size_t)b * N_;

  float m = -3.0e38f;
  for (int i = tid; i < N_ / 4; i += 256) {
    float4 v = row4[i];
    m = fmaxf(m, fmaxf(fmaxf(v.x, v.y), fmaxf(v.z, v.w)));
  }
  red[tid] = m; __syncthreads();
  for (int s = 128; s > 0; s >>= 1) {
    if (tid < s) red[tid] = fmaxf(red[tid], red[tid + s]);
    __syncthreads();
  }
  float rowmax = red[0]; __syncthreads();

  float acc = 0.f;
  for (int i = tid; i < N_ / 4; i += 256) {
    float4 v = row4[i];
    float e0 = __expf(v.x - rowmax), e1 = __expf(v.y - rowmax);
    float e2 = __expf(v.z - rowmax), e3 = __expf(v.w - rowmax);
    v4bf o; o[0] = (bf16)e0; o[1] = (bf16)e1; o[2] = (bf16)e2; o[3] = (bf16)e3;
    *(v4bf*)(orow + i * 4) = o;
    acc += e0 + e1 + e2 + e3;
  }
  red[tid] = acc; __syncthreads();
  for (int s = 128; s > 0; s >>= 1) {
    if (tid < s) red[tid] += red[tid + s];
    __syncthreads();
  }
  if (tid == 0) rowsum[b] = red[0];
}

// ------------------------------------------------------------------
// GEMM2 (NN) + fused eps epilogue:
//   x0[b,d] = (1/rowsum[b]) * sum_n w[b,n] * y[n,d]
//   eps[b,d] = (x[b,d] - sqa[b]*x0[b,d]) * rsv[b]
// WG tile 64(B) x 64(D), 8 waves 4x2, wave 16x32. Double-buffered LDS.
// A tile uses the async path; B tile is transposed through registers.
__global__ __launch_bounds__(256) void k_gemm2(const bf16* __restrict__ pbf,
                                               const bf16* __restrict__ ybf,
                                               const float* __restrict__ rowsum,
                                               const float* __restrict__ sqa,
                                               const float* __restrict__ rsv,
                                               const float* __restrict__ x,
                                               float* __restrict__ out) {
  __shared__ bf16 As[2][64 * LDK];
  __shared__ bf16 Bs[2][64 * LDK];   // transposed: [d_local][k_local]
  const int tid  = threadIdx.x;
  const int m0   = blockIdx.y * 64;   // over B
  const int d0   = blockIdx.x * 64;   // over D
  const int lane = tid & 31, wave = tid >> 5;
  const int wm = wave >> 1, wn = wave & 1;
  const int half = lane >> 4, mm = lane & 15;

  int aRow[2], aCol[2], kR[2], dC[2];
#pragma unroll
  for (int q = 0; q < 2; ++q) {
    int c = tid + q * 256;
    aRow[q] = c >> 3; aCol[q] = (c & 7) * 8;   // A: 64 rows x 8 chunks
    kR[q] = c >> 3;   dC[q] = (c & 7) * 8;     // B: 64 k-rows x 8 chunks of 8 cols
  }

  v8f acc[2] = {};
  v8bf rb[2];
#if !HAVE_ASYNC_LDS
  v8bf ra[2];
#endif

  auto issueCopyA = [&](int buf, int k0) {
#if HAVE_ASYNC_LDS
#pragma unroll
    for (int q = 0; q < 2; ++q)
      async_copy16(&pbf[(size_t)(m0 + aRow[q]) * N_ + k0 + aCol[q]],
                   &As[buf][aRow[q] * LDK + aCol[q]]);
#endif
    (void)buf; (void)k0;
  };
#if !HAVE_ASYNC_LDS
  auto loadA = [&](int k0) {
#pragma unroll
    for (int q = 0; q < 2; ++q)
      ra[q] = *(const v8bf*)&pbf[(size_t)(m0 + aRow[q]) * N_ + k0 + aCol[q]];
  };
  auto storeA = [&](int buf) {
#pragma unroll
    for (int q = 0; q < 2; ++q) *(v8bf*)&As[buf][aRow[q] * LDK + aCol[q]] = ra[q];
  };
#endif
  auto loadB = [&](int k0) {
#pragma unroll
    for (int q = 0; q < 2; ++q)
      rb[q] = *(const v8bf*)&ybf[(size_t)(k0 + kR[q]) * D_ + d0 + dC[q]];
  };
  auto storeB = [&](int buf) {   // transpose 64x64 tile into [d][k]
#pragma unroll
    for (int q = 0; q < 2; ++q)
#pragma unroll
      for (int j = 0; j < 8; ++j)
        Bs[buf][(dC[q] + j) * LDK + kR[q]] = rb[q][j];
  };

  auto compute = [&](int buf) {
#pragma unroll
    for (int kh = 0; kh < 2; ++kh) {
      union { v16bf v; v8bf h[2]; } af;
      const bf16* ap = &As[buf][(wm * 16 + mm) * LDK + kh * 32 + half * 8];
      af.h[0] = *(const v8bf*)ap;
      af.h[1] = *(const v8bf*)(ap + 16);
#pragma unroll
      for (int j = 0; j < 2; ++j) {
        union { v16bf v; v8bf h[2]; } bfr;
        const bf16* bp = &Bs[buf][(wn * 32 + j * 16 + mm) * LDK + kh * 32 + half * 8];
        bfr.h[0] = *(const v8bf*)bp;
        bfr.h[1] = *(const v8bf*)(bp + 16);
        acc[j] = __builtin_amdgcn_wmma_f32_16x16x32_bf16(false, af.v, false, bfr.v,
                                                         (short)0, acc[j], false, false);
      }
    }
  };

  const int nK = N_ / KT;
  // Prologue.
#if HAVE_ASYNC_LDS
  issueCopyA(0, 0);
  loadB(0); storeB(0);
  wait_async0();
#else
  loadA(0); loadB(0);
  storeA(0); storeB(0);
#endif
  __syncthreads();

  for (int i = 0; i < nK; ++i) {
    const int cur = i & 1, nxt = cur ^ 1;
    const int k1 = (i + 1) * KT;
#if HAVE_ASYNC_LDS
    if (i + 1 < nK) { issueCopyA(nxt, k1); loadB(k1); }
    compute(cur);
    if (i + 1 < nK) storeB(nxt);
    wait_async0();
#else
    if (i + 1 < nK) { loadA(k1); loadB(k1); }
    compute(cur);
    if (i + 1 < nK) { storeA(nxt); storeB(nxt); }
#endif
    __syncthreads();
  }

#pragma unroll
  for (int j = 0; j < 2; ++j) {
    int col = d0 + wn * 32 + j * 16 + mm;
#pragma unroll
    for (int i = 0; i < 8; ++i) {
      int row = m0 + wm * 16 + half * 8 + i;
      float x0 = acc[j][i] * (1.0f / rowsum[row]);
      out[(size_t)row * D_ + col] =
          (x[(size_t)row * D_ + col] - sqa[row] * x0) * rsv[row];
    }
  }
}

// ------------------------------------------------------------------
extern "C" void kernel_launch(void* const* d_in, const int* in_sizes, int n_in,
                              void* d_out, int out_size, void* d_ws, size_t ws_size,
                              hipStream_t stream) {
  (void)in_sizes; (void)n_in; (void)out_size; (void)ws_size;
  const float* x  = (const float*)d_in[0];   // [B, D] fp32
  const int*   ts = (const int*)d_in[1];     // [B] int32
  const float* y  = (const float*)d_in[2];   // [N, D] fp32
  float* out = (float*)d_out;                // [B, D] fp32

  char* ws = (char*)d_ws;
  size_t off = 0;
  auto alloc = [&](size_t bytes) -> void* {
    void* p = ws + off;
    off = (off + bytes + 255) & ~(size_t)255;
    return p;
  };
  bf16*  ybf    = (bf16*)alloc((size_t)N_ * D_ * sizeof(bf16));   // 96 MB
  bf16*  xbf    = (bf16*)alloc((size_t)B_ * D_ * sizeof(bf16));   // 1.5 MB
  float* logits = (float*)alloc((size_t)B_ * N_ * sizeof(float)); // 16 MB
  bf16*  pbf    = (bf16*)alloc((size_t)B_ * N_ * sizeof(bf16));   // 8 MB
  float* ysq    = (float*)alloc((size_t)N_ * sizeof(float));
  float* sqa    = (float*)alloc(B_ * sizeof(float));
  float* rsv    = (float*)alloc(B_ * sizeof(float));
  float* c1     = (float*)alloc(B_ * sizeof(float));
  float* c2     = (float*)alloc(B_ * sizeof(float));
  float* varr   = (float*)alloc(B_ * sizeof(float));
  float* c0     = (float*)alloc(B_ * sizeof(float));
  float* rowsum = (float*)alloc(B_ * sizeof(float));

  k_params<<<1, 256, 0, stream>>>(ts, sqa, rsv, c1, c2, varr);
  k_prep_x<<<B_, 256, 0, stream>>>(x, xbf, varr, c0);
  k_prep_y<<<N_, 256, 0, stream>>>(y, ybf, ysq);
  k_gemm1<<<dim3(N_ / 128, B_ / 64), 256, 0, stream>>>(xbf, ybf, ysq, c0, c1, c2, logits);
  k_softmax<<<B_, 256, 0, stream>>>(logits, pbf, rowsum);
  k_gemm2<<<dim3(D_ / 64, B_ / 64), 256, 0, stream>>>(pbf, ybf, rowsum, sqa, rsv, x, out);
}
